// CausalAttention_6811818131543
// MI455X (gfx1250) — matrix operand order repaired and checked by reading, not compiled, not run on hardware
//
#include <hip/hip_runtime.h>
#include <hip/hip_bf16.h>
#include <stdint.h>

typedef __attribute__((ext_vector_type(16))) _Float16 v16h;
typedef __attribute__((ext_vector_type(8)))  _Float16 v8h;
typedef __attribute__((ext_vector_type(8)))  float    v8f;

#define B_  4
#define S_  2048
#define D_  1024

// ---------------------------------------------------------------------------
// f32 -> f16 convert
// ---------------------------------------------------------------------------
__global__ void f32_to_f16_kernel(const float* __restrict__ in,
                                  _Float16* __restrict__ out, int n) {
    int i = blockIdx.x * blockDim.x + threadIdx.x;
    if (i < n) out[i] = (_Float16)in[i];
}

// ---------------------------------------------------------------------------
// Tiled WMMA GEMM: C[M,N] = A[M,K] @ B[K,N], A/B f16 row-major, f32 accumulate.
// Block = 256 threads (8 waves). Tile BM=128 x BN=64, BK=32. Wave w owns rows
// [w*16, w*16+16) x 64 cols (4 wmma n-tiles).
//
// A tile is DMA'd into LDS with CDNA5 async loads (global_load_async_to_lds,
// ASYNCcnt-tracked, no VGPR round trip); B tile is staged transposed so B
// fragments are lane-contiguous.
//
// Fragment layouts used here (wave32):
//  A 16x32: lane (m = l&15, h = l>>4): VGPR0..3 = K h*16..h*16+7,
//           VGPR4..7 = K h*16+8..h*16+15
//  B 32x16: lane (n = l&15, h = l>>4): VGPR0..7 = K h*16 .. h*16+15 (contig, as A; not the ISA table order)
//  C 16x16: lane (n, h): VGPR v -> (row = h*8+v, col = n)
//
// CAUSAL: A = softmax probabilities with A[r, k] == 0 for k > r; reduction
// may stop at k < m0+BM exactly.
// ---------------------------------------------------------------------------
template<bool OUTF32, bool CAUSAL>
__global__ void gemm_wmma_kernel(const _Float16* __restrict__ A,
                                 const _Float16* __restrict__ Bm,
                                 void* __restrict__ Cv,
                                 int K, int lda, int ldb, int ldc,
                                 long sA, long sB, long sC) {
    __shared__ __align__(32) _Float16 lsA[128 * 32];
    __shared__ __align__(32) _Float16 lsBt[64 * 32];

    const _Float16* Ab = A  + (long)blockIdx.z * sA;
    const _Float16* Bb = Bm + (long)blockIdx.z * sB;

    const int t = threadIdx.x;       // 0..255
    const int w = t >> 5;            // wave id 0..7
    const int l = t & 31;            // lane
    const int n = l & 15;            // frag column / A row-in-tile
    const int h = l >> 4;            // half select
    const int m0 = blockIdx.y * 128;
    const int n0 = blockIdx.x * 64;

    const int kEnd = CAUSAL ? (m0 + 128 < K ? m0 + 128 : K) : K;

    v8f acc[4] = {};

    for (int k0 = 0; k0 < kEnd; k0 += 32) {
        // --- async DMA: A tile 128x32 (row-major) into LDS -----------------
        #pragma unroll
        for (int p = 0; p < 2; ++p) {
            int c   = t + p * 256;        // 0..511 chunks of 8 halfs
            int row = c >> 2;
            int kc  = (c & 3) * 8;
            unsigned ldsoff =
                (unsigned)(uintptr_t)&lsA[row * 32 + kc];
            unsigned long long ga = (unsigned long long)(uintptr_t)
                (Ab + (long)(m0 + row) * lda + k0 + kc);
            asm volatile("global_load_async_to_lds_b128 %0, %1, off"
                         :: "v"(ldsoff), "v"(ga) : "memory");
        }
        // --- B tile 32x64 staged transposed -> lsBt[n][k] ------------------
        {
            int kr = t >> 3;              // 0..31
            int nc = (t & 7) * 8;
            v8h bv = *(const v8h*)(Bb + (long)(k0 + kr) * ldb + n0 + nc);
            #pragma unroll
            for (int j = 0; j < 8; ++j) lsBt[(nc + j) * 32 + kr] = bv[j];
        }
        asm volatile("s_wait_asynccnt 0x0" ::: "memory");
        __syncthreads();

        v16h a;
        {
            const _Float16* ar = &lsA[(w * 16 + n) * 32];
            v8h lo = *(const v8h*)(ar + h * 16);
            v8h hi = *(const v8h*)(ar + h * 16 + 8);
            #pragma unroll
            for (int i = 0; i < 8; ++i) { a[i] = lo[i]; a[i + 8] = hi[i]; }
        }
        #pragma unroll
        for (int t4 = 0; t4 < 4; ++t4) {
            v16h bb = *(const v16h*)&lsBt[(t4 * 16 + n) * 32 + h * 16];
            acc[t4] = __builtin_amdgcn_wmma_f32_16x16x32_f16(
                false, a, false, bb, (short)0, acc[t4], false, false);
        }
        __syncthreads();
    }

    #pragma unroll
    for (int t4 = 0; t4 < 4; ++t4)
        #pragma unroll
        for (int v = 0; v < 8; ++v) {
            long row = m0 + w * 16 + h * 8 + v;
            long col = n0 + t4 * 16 + n;
            float val = acc[t4][v];
            if (OUTF32)
                ((float*)Cv)[(long)blockIdx.z * sC + row * ldc + col] = val;
            else
                ((_Float16*)Cv)[(long)blockIdx.z * sC + row * ldc + col] =
                    (_Float16)val;
        }
}

// ---------------------------------------------------------------------------
// Scores: P_raw[b,q,k] = (Q[b,q,:] . K[b,k,:]) / 32, causal mask -> -30000.
// NT GEMM: both A (Q rows) and B (K rows as columns of K^T) load contiguously
// from global (whole f16 working set is L2-resident). One wave = 16q x 64k
// tile over full D=1024 reduction. Fully masked tiles skip all compute.
// Block = 128 threads = 4 waves, each wave a different 64-wide key strip.
// ---------------------------------------------------------------------------
__global__ void scores_wmma_kernel(const _Float16* __restrict__ Qh,
                                   const _Float16* __restrict__ Kh,
                                   _Float16* __restrict__ P) {
    const int w  = threadIdx.x >> 5;
    const int l  = threadIdx.x & 31;
    const int n  = l & 15;
    const int h  = l >> 4;
    const int qt = blockIdx.x;              // 0 .. B*S/16-1
    const int b  = qt >> 7;                 // S/16 = 128 q-tiles per batch
    const int q0 = (qt & 127) << 4;
    const int k0 = (blockIdx.y << 8) + (w << 6);

    v8f acc[4] = {};
    if (k0 <= q0 + 15) {                    // tile intersects causal region
        const _Float16* qrow = Qh + (long)(b * S_ + q0 + n) * D_;
        for (int d0 = 0; d0 < D_; d0 += 32) {
            v16h a;
            v8h lo = *(const v8h*)(qrow + d0 + h * 16);
            v8h hi = *(const v8h*)(qrow + d0 + h * 16 + 8);
            #pragma unroll
            for (int i = 0; i < 8; ++i) { a[i] = lo[i]; a[i + 8] = hi[i]; }
            #pragma unroll
            for (int t4 = 0; t4 < 4; ++t4) {
                const _Float16* krow =
                    Kh + (long)(b * S_ + k0 + t4 * 16 + n) * D_;
                v16h bb = *(const v16h*)(krow + d0 + h * 16);
                acc[t4] = __builtin_amdgcn_wmma_f32_16x16x32_f16(
                    false, a, false, bb, (short)0, acc[t4], false, false);
            }
        }
    }
    const float scale = 0.03125f;           // 1/sqrt(1024)
    #pragma unroll
    for (int t4 = 0; t4 < 4; ++t4)
        #pragma unroll
        for (int v = 0; v < 8; ++v) {
            int row = q0 + h * 8 + v;
            int col = k0 + t4 * 16 + n;
            float val = (col <= row) ? acc[t4][v] * scale : -30000.0f;
            P[((long)b * S_ + row) * S_ + col] = (_Float16)val;
        }
}

// ---------------------------------------------------------------------------
// Row softmax, in-place on f16 scores -> probabilities. One 256-thread block
// per row (8192 rows), 8 elements/thread, f32 math. Masked entries (-30000)
// underflow to exactly 0 after exp, so the causal P.V skip is exact.
// ---------------------------------------------------------------------------
__global__ void softmax_rows_kernel(_Float16* __restrict__ P) {
    __shared__ float red[256];
    const long row = blockIdx.x;
    _Float16* p = P + row * (long)S_;
    const int t = threadIdx.x;

    float v[8];
    float m = -3.0e38f;
    #pragma unroll
    for (int i = 0; i < 8; ++i) {
        v[i] = (float)p[t + i * 256];
        m = fmaxf(m, v[i]);
    }
    red[t] = m; __syncthreads();
    for (int s = 128; s > 0; s >>= 1) {
        if (t < s) red[t] = fmaxf(red[t], red[t + s]);
        __syncthreads();
    }
    m = red[0]; __syncthreads();

    float sum = 0.f;
    #pragma unroll
    for (int i = 0; i < 8; ++i) { v[i] = expf(v[i] - m); sum += v[i]; }
    red[t] = sum; __syncthreads();
    for (int s = 128; s > 0; s >>= 1) {
        if (t < s) red[t] += red[t + s];
        __syncthreads();
    }
    float inv = 1.0f / red[0];
    #pragma unroll
    for (int i = 0; i < 8; ++i)
        p[t + i * 256] = (_Float16)(v[i] * inv);
}

// ---------------------------------------------------------------------------
// Launcher.  Workspace layout (f16 elements):
//   xh  [8192,1024] | wqh/wkh/wvh [1024,1024]x3 | Qh | Kh | Vh [8192,1024]x3
//   P   [4,2048,2048]   (scores, then probabilities, in place)
// Total ~107 MB — L2-resident on MI455X (192 MB).
// ---------------------------------------------------------------------------
extern "C" void kernel_launch(void* const* d_in, const int* in_sizes, int n_in,
                              void* d_out, int out_size, void* d_ws,
                              size_t ws_size, hipStream_t stream) {
    (void)in_sizes; (void)n_in; (void)out_size; (void)ws_size;
    const float* x  = (const float*)d_in[0];
    const float* wq = (const float*)d_in[1];
    const float* wk = (const float*)d_in[2];
    const float* wv = (const float*)d_in[3];
    float* out = (float*)d_out;

    const long MX = (long)B_ * S_;           // 8192
    _Float16* xh  = (_Float16*)d_ws;
    _Float16* wqh = xh  + MX * D_;
    _Float16* wkh = wqh + (long)D_ * D_;
    _Float16* wvh = wkh + (long)D_ * D_;
    _Float16* Qh  = wvh + (long)D_ * D_;
    _Float16* Kh  = Qh  + MX * D_;
    _Float16* Vh  = Kh  + MX * D_;
    _Float16* Pp  = Vh  + MX * D_;           // [B,S,S]

    // 1) precision downconvert
    {
        int nx = (int)(MX * D_), nw = D_ * D_;
        f32_to_f16_kernel<<<(nx + 255) / 256, 256, 0, stream>>>(x,  xh,  nx);
        f32_to_f16_kernel<<<(nw + 255) / 256, 256, 0, stream>>>(wq, wqh, nw);
        f32_to_f16_kernel<<<(nw + 255) / 256, 256, 0, stream>>>(wk, wkh, nw);
        f32_to_f16_kernel<<<(nw + 255) / 256, 256, 0, stream>>>(wv, wvh, nw);
    }

    // 2) QKV projections: [8192,1024] @ [1024,1024] -> f16
    {
        dim3 grid(D_ / 64, (unsigned)(MX / 128), 1), blk(256);
        gemm_wmma_kernel<false, false><<<grid, blk, 0, stream>>>(
            xh, wqh, Qh, D_, D_, D_, D_, 0, 0, 0);
        gemm_wmma_kernel<false, false><<<grid, blk, 0, stream>>>(
            xh, wkh, Kh, D_, D_, D_, D_, 0, 0, 0);
        gemm_wmma_kernel<false, false><<<grid, blk, 0, stream>>>(
            xh, wvh, Vh, D_, D_, D_, D_, 0, 0, 0);
    }

    // 3) causal scaled scores (f16), Q.K^T
    scores_wmma_kernel<<<dim3((unsigned)(MX / 16), S_ / 256, 1), dim3(128),
                         0, stream>>>(Qh, Kh, Pp);

    // 4) row softmax in place
    softmax_rows_kernel<<<dim3((unsigned)MX), dim3(256), 0, stream>>>(Pp);

    // 5) O = P @ V per batch -> f32 output (causal: skip k-blocks past m0+BM)
    gemm_wmma_kernel<true, true><<<dim3(D_ / 64, S_ / 128, B_), dim3(256), 0,
                                   stream>>>(
        Pp, Vh, out, S_, S_, D_, D_,
        (long)S_ * S_, (long)S_ * D_, (long)S_ * D_);
}
